// RelMultiHeadAttn_52793738002677
// MI455X (gfx1250) — matrix operand order repaired and checked by reading, not compiled, not run on hardware
//
#include <hip/hip_runtime.h>
#include <hip/hip_bf16.h>

// ---------------------------------------------------------------------------
// Problem constants (from reference)
// ---------------------------------------------------------------------------
#define QLEN   2048
#define RLEN   2049
#define BSZ    2
#define DMODEL 1024
#define NHEAD  16
#define DHEAD  64
#define NQKV   3072          // 3 * NHEAD * DHEAD
#define MROWS  (QLEN * BSZ)  // 4096 flattened (i,b) rows

typedef __attribute__((ext_vector_type(16))) __bf16 v16bf;
typedef __attribute__((ext_vector_type(8)))  float  v8f;

// ---------------------------------------------------------------------------
// bf16 <-> f32 helpers (bit-exact RNE, no reliance on __bf16 arithmetic)
// ---------------------------------------------------------------------------
union BFU { unsigned short u; __bf16 b; };

static __device__ __forceinline__ __bf16 f2bf(float f) {
  unsigned x = __builtin_bit_cast(unsigned, f);
  unsigned r = (x + 0x7FFFu + ((x >> 16) & 1u)) >> 16;
  BFU t; t.u = (unsigned short)r; return t.b;
}
static __device__ __forceinline__ float bf2f(__bf16 b) {
  BFU t; t.b = b;
  unsigned x = ((unsigned)t.u) << 16;
  return __builtin_bit_cast(float, x);
}
static __device__ __forceinline__ v8f vzero8() {
  v8f z;
#pragma unroll
  for (int i = 0; i < 8; ++i) z[i] = 0.f;
  return z;
}

// ---------------------------------------------------------------------------
// WMMA fragment loaders (wave32 layouts per CDNA5 ISA 7.12.2)
//   A 16x32 bf16 : lane (m=lane&15, lh=lane>>4), element e -> k=(e&7)+(e>>3)*16+8*lh
//   B 32x16 bf16 : lane (n=lane&15, lh=lane>>4), element e -> k=(e&7)+(e>>3)*16+8*lh
//   C/D 16x16 f32: vgpr p, lane -> (row=p+8*lh, col=lane&15)
// ---------------------------------------------------------------------------
static __device__ __forceinline__ v16bf load_A(const __bf16* src, int ldm,
                                               int m0, int k0, int lane) {
  int m  = m0 + (lane & 15);
  int lh = (lane >> 4) & 1;
  const __bf16* p = src + (size_t)m * ldm + k0 + 8 * lh;
  v16bf a;
#pragma unroll
  for (int e = 0; e < 8; ++e) a[e]     = p[e];
#pragma unroll
  for (int e = 0; e < 8; ++e) a[8 + e] = p[16 + e];
  return a;
}

static __device__ __forceinline__ v16bf load_A_clamp(const __bf16* src, int ldm,
                                                     int m0, int k0, int lane, int mmax) {
  int m  = m0 + (lane & 15);
  if (m > mmax) m = mmax;
  int lh = (lane >> 4) & 1;
  const __bf16* p = src + (size_t)m * ldm + k0 + 8 * lh;
  v16bf a;
#pragma unroll
  for (int e = 0; e < 8; ++e) a[e]     = p[e];
#pragma unroll
  for (int e = 0; e < 8; ++e) a[8 + e] = p[16 + e];
  return a;
}

// B = src^T where src is row-major [N][K]: B[k][n] = src[n0+n][k0+k]
static __device__ __forceinline__ v16bf load_Bt(const __bf16* src, int ldn,
                                                int n0, int k0, int lane) {
  int n  = n0 + (lane & 15);
  int lh = (lane >> 4) & 1;
  const __bf16* p = src + (size_t)n * ldn + k0 + 8 * lh;
  v16bf b;
#pragma unroll
  for (int e = 0; e < 8; ++e) b[e]     = p[e];
#pragma unroll
  for (int e = 0; e < 8; ++e) b[8 + e] = p[16 + e];
  return b;
}

static __device__ __forceinline__ v16bf load_Bt_clamp(const __bf16* src, int ldn,
                                                      int n0, int k0, int lane, int nmax) {
  int n  = n0 + (lane & 15);
  if (n > nmax) n = nmax;
  int lh = (lane >> 4) & 1;
  const __bf16* p = src + (size_t)n * ldn + k0 + 8 * lh;
  v16bf b;
#pragma unroll
  for (int e = 0; e < 8; ++e) b[e]     = p[e];
#pragma unroll
  for (int e = 0; e < 8; ++e) b[8 + e] = p[16 + e];
  return b;
}

static __device__ __forceinline__ v8f wmma_bf16(v16bf a, v16bf b, v8f c) {
  return __builtin_amdgcn_wmma_f32_16x16x32_bf16(false, a, false, b,
                                                 (short)0, c, false, false);
}

// ---------------------------------------------------------------------------
// Software-pipelined 32x64 per-wave GEMM tile with ping-pong double buffering
// (k-loop unrolled x2: two fragment sets alternate load/consume, so no
// register-rotation copies and no WMMA->VALU hazard NOPs).  Requires K % 64 == 0.
// acc[rg*4+nc]: rg = 16-row group, nc = 16-col group.
// ---------------------------------------------------------------------------
static __device__ __forceinline__ void gemm_tile_32x64(
    const __bf16* __restrict__ A, int lda, int m0, int mmax,
    const __bf16* __restrict__ B, int ldb, int n0,
    int K, int lane, v8f acc[8]) {
  v16bf a0 = load_A_clamp(A, lda, m0,      0, lane, mmax);
  v16bf a1 = load_A_clamp(A, lda, m0 + 16, 0, lane, mmax);
  v16bf b0 = load_Bt(B, ldb, n0,      0, lane);
  v16bf b1 = load_Bt(B, ldb, n0 + 16, 0, lane);
  v16bf b2 = load_Bt(B, ldb, n0 + 32, 0, lane);
  v16bf b3 = load_Bt(B, ldb, n0 + 48, 0, lane);
  for (int k0 = 0; k0 < K; k0 += 64) {
    int k1 = k0 + 32;                          // always < K (K % 64 == 0)
    v16bf c0 = load_A_clamp(A, lda, m0,      k1, lane, mmax);
    v16bf c1 = load_A_clamp(A, lda, m0 + 16, k1, lane, mmax);
    v16bf d0 = load_Bt(B, ldb, n0,      k1, lane);
    v16bf d1 = load_Bt(B, ldb, n0 + 16, k1, lane);
    v16bf d2 = load_Bt(B, ldb, n0 + 32, k1, lane);
    v16bf d3 = load_Bt(B, ldb, n0 + 48, k1, lane);
    if (k0 + 128 < K)
      __builtin_prefetch(B + (size_t)(n0 + (lane & 15)) * ldb + k0 + 128, 0, 1);
    acc[0] = wmma_bf16(a0, b0, acc[0]);
    acc[1] = wmma_bf16(a0, b1, acc[1]);
    acc[2] = wmma_bf16(a0, b2, acc[2]);
    acc[3] = wmma_bf16(a0, b3, acc[3]);
    acc[4] = wmma_bf16(a1, b0, acc[4]);
    acc[5] = wmma_bf16(a1, b1, acc[5]);
    acc[6] = wmma_bf16(a1, b2, acc[6]);
    acc[7] = wmma_bf16(a1, b3, acc[7]);
    int k2 = (k0 + 64 < K) ? k0 + 64 : 0;      // last iter: harmless reload
    a0 = load_A_clamp(A, lda, m0,      k2, lane, mmax);
    a1 = load_A_clamp(A, lda, m0 + 16, k2, lane, mmax);
    b0 = load_Bt(B, ldb, n0,      k2, lane);
    b1 = load_Bt(B, ldb, n0 + 16, k2, lane);
    b2 = load_Bt(B, ldb, n0 + 32, k2, lane);
    b3 = load_Bt(B, ldb, n0 + 48, k2, lane);
    acc[0] = wmma_bf16(c0, d0, acc[0]);
    acc[1] = wmma_bf16(c0, d1, acc[1]);
    acc[2] = wmma_bf16(c0, d2, acc[2]);
    acc[3] = wmma_bf16(c0, d3, acc[3]);
    acc[4] = wmma_bf16(c1, d0, acc[4]);
    acc[5] = wmma_bf16(c1, d1, acc[5]);
    acc[6] = wmma_bf16(c1, d2, acc[6]);
    acc[7] = wmma_bf16(c1, d3, acc[7]);
  }
}

// ---------------------------------------------------------------------------
// Kernel 0: f32 -> bf16 conversion (grid-stride)
// ---------------------------------------------------------------------------
__global__ void k_cvt(const float* __restrict__ in, __bf16* __restrict__ out, int n) {
  for (int i = blockIdx.x * blockDim.x + threadIdx.x; i < n;
       i += gridDim.x * blockDim.x)
    out[i] = f2bf(in[i]);
}

// ---------------------------------------------------------------------------
// Kernel 1: r projection. r_bf[2049][1024] @ W_qkv^T, columns 0..2047.
//  t==1 third  -> rk[h][jr][d]  (bf16)
//  t==0 third, row 2048 -> rq_last[1024] (f32)   (= r_q[-1] bias)
// ---------------------------------------------------------------------------
__global__ void __launch_bounds__(128, 1)
k_gemm_r(const __bf16* __restrict__ rbf,
         const __bf16* __restrict__ wqkv,
         __bf16* __restrict__ rk,
         float* __restrict__ rq_last) {
  int lane = threadIdx.x & 31;
  int wave = threadIdx.x >> 5;
  int m0 = blockIdx.x * 128 + wave * 32;
  int n0 = blockIdx.y * 64;

  v8f acc[8];
#pragma unroll
  for (int i = 0; i < 8; ++i) acc[i] = vzero8();
  gemm_tile_32x64(rbf, DMODEL, m0, RLEN - 1, wqkv, DMODEL, n0, DMODEL, lane, acc);

  int lh = lane >> 4, nl = lane & 15;
#pragma unroll
  for (int rg = 0; rg < 2; ++rg) {
#pragma unroll
    for (int nc = 0; nc < 4; ++nc) {
#pragma unroll
      for (int p = 0; p < 8; ++p) {
        int m = m0 + rg * 16 + p + 8 * lh;
        if (m >= RLEN) continue;
        int c = n0 + nc * 16 + nl;
        int t = c >> 10, cc = c & 1023;
        float v = acc[rg * 4 + nc][p];
        if (t == 1) {
          int h = cc >> 6, d = cc & 63;
          rk[((size_t)h * RLEN + m) * DHEAD + d] = f2bf(v);
        } else if (m == RLEN - 1) {
          rq_last[cc] = v;
        }
      }
    }
  }
}

// ---------------------------------------------------------------------------
// Kernel 2: QKV projection. w_bf[4096][1024] @ W_qkv^T -> Q(+bias)/K/V^T
// ---------------------------------------------------------------------------
__global__ void __launch_bounds__(128, 1)
k_gemm_qkv(const __bf16* __restrict__ wbf,
           const __bf16* __restrict__ wqkv,
           const float* __restrict__ rq_last,
           __bf16* __restrict__ Qb,
           __bf16* __restrict__ Kb,
           __bf16* __restrict__ Vtb) {
  int lane = threadIdx.x & 31;
  int wave = threadIdx.x >> 5;
  int m0 = blockIdx.x * 128 + wave * 32;
  int n0 = blockIdx.y * 64;

  v8f acc[8];
#pragma unroll
  for (int i = 0; i < 8; ++i) acc[i] = vzero8();
  gemm_tile_32x64(wbf, DMODEL, m0, MROWS - 1, wqkv, DMODEL, n0, DMODEL, lane, acc);

  int lh = lane >> 4, nl = lane & 15;
#pragma unroll
  for (int rg = 0; rg < 2; ++rg) {
#pragma unroll
    for (int nc = 0; nc < 4; ++nc) {
#pragma unroll
      for (int p = 0; p < 8; ++p) {
        int r = m0 + rg * 16 + p + 8 * lh;   // flat (i,b) row of w
        int i = r >> 1, bb = r & 1;
        int c = n0 + nc * 16 + nl;
        int t = c >> 10, cc = c & 1023;
        int h = cc >> 6, d = cc & 63;
        float v = acc[rg * 4 + nc][p];
        size_t bh = (size_t)bb * NHEAD + h;
        if (t == 0) {
          Qb[(bh * QLEN + i) * DHEAD + d] = f2bf(v + rq_last[cc]);   // rw_q
        } else if (t == 1) {
          Kb[(bh * QLEN + i) * DHEAD + d] = f2bf(v);
        } else {
          Vtb[(bh * DHEAD + d) * QLEN + i] = f2bf(v);                // transposed V
        }
      }
    }
  }
}

// ---------------------------------------------------------------------------
// Kernel 3: attention. One wave per (b,h, 16-query tile). Flash-style online
// softmax over k tiles of 32 (loop ping-ponged x2). BD (rel-shift) precomputed
// per q-tile into LDS; K fragment loads hidden under the softmax VALU section.
// ---------------------------------------------------------------------------
#define LDB   2080                     // padded BD row (>= RLEN + 31 for ping-pong tail)
#define WLDSE (17 * LDB + 16 * 32)     // per-wave LDS elements (BD rows + P scratch)

__global__ void __launch_bounds__(64, 1)
k_attn(const __bf16* __restrict__ Qb,
       const __bf16* __restrict__ Kb,
       const __bf16* __restrict__ Vtb,
       const __bf16* __restrict__ rk,
       __bf16* __restrict__ av) {
  extern __shared__ char smem[];
  int lane = threadIdx.x & 31;
  int wave = threadIdx.x >> 5;
  __bf16* BD = (__bf16*)smem + (size_t)wave * WLDSE;
  __bf16* Ps = BD + 17 * LDB;

  int i0 = (blockIdx.x * 2 + wave) * 16;
  int bh = blockIdx.y;                 // b*16 + h
  int h  = bh & 15;
  int bb = bh >> 4;
  const __bf16* Qh = Qb  + (size_t)bh * QLEN * DHEAD;
  const __bf16* Kh = Kb  + (size_t)bh * QLEN * DHEAD;
  const __bf16* Vh = Vtb + (size_t)bh * DHEAD * QLEN;
  const __bf16* Rh = rk  + (size_t)h * RLEN * DHEAD;
  int lh = lane >> 4, nl = lane & 15;

  // Q fragments for this 16-row tile (reused by BD, AC)
  v16bf aq0 = load_A(Qh, DHEAD, i0, 0,  lane);
  v16bf aq1 = load_A(Qh, DHEAD, i0, 32, lane);

  // ---- BD rows 0..15 for all 2049 relative positions (WMMA, ping-ponged) ----
  {
    v16bf rb0 = load_Bt_clamp(Rh, DHEAD, 0, 0,  lane, RLEN - 1);
    v16bf rb1 = load_Bt_clamp(Rh, DHEAD, 0, 32, lane, RLEN - 1);
    for (int jr0 = 0; jr0 < RLEN; jr0 += 32) {
      int j1 = jr0 + 16;                          // may exceed RLEN-1: clamped
      v16bf sb0 = load_Bt_clamp(Rh, DHEAD, j1, 0,  lane, RLEN - 1);
      v16bf sb1 = load_Bt_clamp(Rh, DHEAD, j1, 32, lane, RLEN - 1);
      v8f acc = vzero8();
      acc = wmma_bf16(aq0, rb0, acc);
      acc = wmma_bf16(aq1, rb1, acc);
#pragma unroll
      for (int p = 0; p < 8; ++p)
        BD[(p + 8 * lh) * LDB + jr0 + nl] = f2bf(acc[p]);
      int j2 = (jr0 + 32 < RLEN) ? jr0 + 32 : 0;  // last iter: harmless reload
      rb0 = load_Bt_clamp(Rh, DHEAD, j2, 0,  lane, RLEN - 1);
      rb1 = load_Bt_clamp(Rh, DHEAD, j2, 32, lane, RLEN - 1);
      v8f acc2 = vzero8();
      acc2 = wmma_bf16(aq0, sb0, acc2);
      acc2 = wmma_bf16(aq1, sb1, acc2);
#pragma unroll
      for (int p = 0; p < 8; ++p)
        BD[(p + 8 * lh) * LDB + j1 + nl] = f2bf(acc2[p]);   // tail cols < LDB
    }
  }
  // ---- BD extra row: query i0+16 (needed by rel-shift wrap, VALU) ----
  if (i0 + 16 < QLEN) {
    for (int jr = lane; jr < RLEN; jr += 32) {
      float s = 0.f;
#pragma unroll 8
      for (int d = 0; d < DHEAD; ++d)
        s += bf2f(Qh[(size_t)(i0 + 16) * DHEAD + d]) * bf2f(Rh[(size_t)jr * DHEAD + d]);
      BD[16 * LDB + jr] = f2bf(s);
    }
  }

  // ---- flash loop over keys (ping-ponged x2: 64 keys per iteration) ----
  float mrow[8], lrow[8];
#pragma unroll
  for (int p = 0; p < 8; ++p) { mrow[p] = -3.0e38f; lrow[p] = 0.f; }
  v8f Oacc[4];
#pragma unroll
  for (int nc = 0; nc < 4; ++nc) Oacc[nc] = vzero8();

  // one 32-key phase: scores, rel-shift gather, online softmax, P@V
  auto phase = [&](int jj, v16bf pk0, v16bf pk1, v16bf pk2, v16bf pk3) {
    v16bf bv0 = load_Bt(Vh, QLEN, 0,  jj, lane);
    v16bf bv1 = load_Bt(Vh, QLEN, 16, jj, lane);
    v16bf bv2 = load_Bt(Vh, QLEN, 32, jj, lane);
    v16bf bv3 = load_Bt(Vh, QLEN, 48, jj, lane);

    v8f s0 = vzero8(), s1 = vzero8();
    s0 = wmma_bf16(aq0, pk0, s0);
    s0 = wmma_bf16(aq1, pk1, s0);
    s1 = wmma_bf16(aq0, pk2, s1);
    s1 = wmma_bf16(aq1, pk3, s1);

    float alpha[8];
#pragma unroll
    for (int p = 0; p < 8; ++p) {
      int ti = p + 8 * lh;
      int qi = i0 + ti;
      int jc0 = jj + nl, jc1 = jj + 16 + nl;
      // rel-shift gather: j<=i -> BD[ti][j+QLEN-i] ; j>i -> BD[ti+1][j-i-1]
      int r0 = (jc0 <= qi) ? ti : ti + 1;
      int o0 = (jc0 <= qi) ? (jc0 + QLEN - qi) : (jc0 - qi - 1);
      int r1 = (jc1 <= qi) ? ti : ti + 1;
      int o1 = (jc1 <= qi) ? (jc1 + QLEN - qi) : (jc1 - qi - 1);
      float v0 = (s0[p] + bf2f(BD[r0 * LDB + o0])) * 0.125f;
      float v1 = (s1[p] + bf2f(BD[r1 * LDB + o1])) * 0.125f;

      float mx = fmaxf(v0, v1);
      mx = fmaxf(mx, __shfl_xor(mx, 1, 32));
      mx = fmaxf(mx, __shfl_xor(mx, 2, 32));
      mx = fmaxf(mx, __shfl_xor(mx, 4, 32));
      mx = fmaxf(mx, __shfl_xor(mx, 8, 32));
      float mnew = fmaxf(mrow[p], mx);
      float al = __expf(mrow[p] - mnew);
      float e0 = __expf(v0 - mnew);
      float e1 = __expf(v1 - mnew);
      float ps = e0 + e1;
      ps += __shfl_xor(ps, 1, 32);
      ps += __shfl_xor(ps, 2, 32);
      ps += __shfl_xor(ps, 4, 32);
      ps += __shfl_xor(ps, 8, 32);
      lrow[p] = lrow[p] * al + ps;
      mrow[p] = mnew;
      alpha[p] = al;
      Ps[ti * 32 + nl]      = f2bf(e0);
      Ps[ti * 32 + 16 + nl] = f2bf(e1);
    }

    v16bf ap = load_A((const __bf16*)Ps, 32, 0, 0, lane);
#pragma unroll
    for (int nc = 0; nc < 4; ++nc) {
#pragma unroll
      for (int p = 0; p < 8; ++p) Oacc[nc][p] *= alpha[p];
    }
    Oacc[0] = wmma_bf16(ap, bv0, Oacc[0]);
    Oacc[1] = wmma_bf16(ap, bv1, Oacc[1]);
    Oacc[2] = wmma_bf16(ap, bv2, Oacc[2]);
    Oacc[3] = wmma_bf16(ap, bv3, Oacc[3]);
  };

  v16bf bk0 = load_Bt(Kh, DHEAD, 0,  0,  lane);
  v16bf bk1 = load_Bt(Kh, DHEAD, 0,  32, lane);
  v16bf bk2 = load_Bt(Kh, DHEAD, 16, 0,  lane);
  v16bf bk3 = load_Bt(Kh, DHEAD, 16, 32, lane);

  for (int j0 = 0; j0 < QLEN; j0 += 64) {
    int j1 = j0 + 32;                            // always < QLEN (QLEN % 64 == 0)
    v16bf ck0 = load_Bt(Kh, DHEAD, j1,      0,  lane);
    v16bf ck1 = load_Bt(Kh, DHEAD, j1,      32, lane);
    v16bf ck2 = load_Bt(Kh, DHEAD, j1 + 16, 0,  lane);
    v16bf ck3 = load_Bt(Kh, DHEAD, j1 + 16, 32, lane);
    phase(j0, bk0, bk1, bk2, bk3);
    int j2 = (j0 + 64 < QLEN) ? j0 + 64 : 0;     // last iter: harmless reload
    bk0 = load_Bt(Kh, DHEAD, j2,      0,  lane);
    bk1 = load_Bt(Kh, DHEAD, j2,      32, lane);
    bk2 = load_Bt(Kh, DHEAD, j2 + 16, 0,  lane);
    bk3 = load_Bt(Kh, DHEAD, j2 + 16, 32, lane);
    phase(j1, ck0, ck1, ck2, ck3);
  }

  // ---- write attn_vec rows [(i*BSZ+b)][h*64 + d] (bf16) ----
#pragma unroll
  for (int nc = 0; nc < 4; ++nc) {
#pragma unroll
    for (int p = 0; p < 8; ++p) {
      int ti = p + 8 * lh;
      int qi = i0 + ti;
      int row = qi * BSZ + bb;
      int col = h * DHEAD + nc * 16 + nl;
      av[(size_t)row * DMODEL + col] = f2bf(Oacc[nc][p] / lrow[p]);
    }
  }
}

// ---------------------------------------------------------------------------
// Kernel 4: output projection + residual. x = attn_vec @ W_o^T + w
// ---------------------------------------------------------------------------
__global__ void __launch_bounds__(128, 1)
k_gemm_out(const __bf16* __restrict__ avb,
           const __bf16* __restrict__ wob,
           const float* __restrict__ w_in,
           float* __restrict__ xb) {
  int lane = threadIdx.x & 31;
  int wave = threadIdx.x >> 5;
  int m0 = blockIdx.x * 128 + wave * 32;
  int n0 = blockIdx.y * 64;

  v8f acc[8];
#pragma unroll
  for (int i = 0; i < 8; ++i) acc[i] = vzero8();
  gemm_tile_32x64(avb, DMODEL, m0, MROWS - 1, wob, DMODEL, n0, DMODEL, lane, acc);

  int lh = lane >> 4, nl = lane & 15;
#pragma unroll
  for (int rg = 0; rg < 2; ++rg) {
#pragma unroll
    for (int nc = 0; nc < 4; ++nc) {
#pragma unroll
      for (int p = 0; p < 8; ++p) {
        int r = m0 + rg * 16 + p + 8 * lh;
        int c = n0 + nc * 16 + nl;
        xb[(size_t)r * DMODEL + c] = acc[rg * 4 + nc][p] + w_in[(size_t)r * DMODEL + c];
      }
    }
  }
}

// ---------------------------------------------------------------------------
// Kernel 5: LayerNorm over d_model per row
// ---------------------------------------------------------------------------
__global__ void k_ln(const float* __restrict__ xb,
                     const float* __restrict__ gamma,
                     const float* __restrict__ beta,
                     float* __restrict__ out) {
  int row = blockIdx.x;
  const float* x = xb + (size_t)row * DMODEL;
  float s = 0.f, sq = 0.f;
  for (int c = threadIdx.x; c < DMODEL; c += blockDim.x) {
    float v = x[c]; s += v; sq += v * v;
  }
  for (int m = 16; m >= 1; m >>= 1) {
    s  += __shfl_xor(s,  m, 32);
    sq += __shfl_xor(sq, m, 32);
  }
  __shared__ float red[2][8];
  int lane = threadIdx.x & 31, wv = threadIdx.x >> 5;
  if (lane == 0) { red[0][wv] = s; red[1][wv] = sq; }
  __syncthreads();
  float ts = 0.f, tsq = 0.f;
#pragma unroll
  for (int i = 0; i < 8; ++i) { ts += red[0][i]; tsq += red[1][i]; }
  float mu  = ts  * (1.0f / DMODEL);
  float var = tsq * (1.0f / DMODEL) - mu * mu;
  float inv = rsqrtf(var + 1e-5f);
  for (int c = threadIdx.x; c < DMODEL; c += blockDim.x)
    out[(size_t)row * DMODEL + c] = (x[c] - mu) * inv * gamma[c] + beta[c];
}

// ---------------------------------------------------------------------------
// Host launcher
// ---------------------------------------------------------------------------
static inline char* ws_take(char*& p, size_t bytes) {
  char* r = p;
  p += (bytes + 255) & ~(size_t)255;
  return r;
}

extern "C" void kernel_launch(void* const* d_in, const int* in_sizes, int n_in,
                              void* d_out, int out_size, void* d_ws, size_t ws_size,
                              hipStream_t stream) {
  const float* w     = (const float*)d_in[0];   // [2048, 2, 1024]
  const float* r     = (const float*)d_in[1];   // [2049, 1024]
  const float* Wqkv  = (const float*)d_in[2];   // [3072, 1024]
  const float* Wo    = (const float*)d_in[3];   // [1024, 1024]
  const float* gamma = (const float*)d_in[4];   // [1024]
  const float* beta  = (const float*)d_in[5];   // [1024]
  (void)in_sizes; (void)n_in; (void)out_size; (void)ws_size;

  char* p = (char*)d_ws;
  __bf16* wqkv_bf = (__bf16*)ws_take(p, (size_t)NQKV * DMODEL * 2);
  __bf16* wo_bf   = (__bf16*)ws_take(p, (size_t)DMODEL * DMODEL * 2);
  __bf16* w_bf    = (__bf16*)ws_take(p, (size_t)MROWS * DMODEL * 2);
  __bf16* r_bf    = (__bf16*)ws_take(p, (size_t)RLEN * DMODEL * 2);
  __bf16* rk      = (__bf16*)ws_take(p, (size_t)NHEAD * RLEN * DHEAD * 2);
  float*  rq_last = (float*) ws_take(p, (size_t)DMODEL * 4);
  __bf16* Qb      = (__bf16*)ws_take(p, (size_t)BSZ * NHEAD * QLEN * DHEAD * 2);
  __bf16* Kb      = (__bf16*)ws_take(p, (size_t)BSZ * NHEAD * QLEN * DHEAD * 2);
  __bf16* Vtb     = (__bf16*)ws_take(p, (size_t)BSZ * NHEAD * DHEAD * QLEN * 2);
  __bf16* avb     = (__bf16*)ws_take(p, (size_t)MROWS * DMODEL * 2);
  float*  xb      = (float*) ws_take(p, (size_t)MROWS * DMODEL * 4);

  // 0) convert all f32 operands to bf16 once
  k_cvt<<<4096, 256, 0, stream>>>(Wqkv, wqkv_bf, NQKV * DMODEL);
  k_cvt<<<4096, 256, 0, stream>>>(Wo,   wo_bf,   DMODEL * DMODEL);
  k_cvt<<<4096, 256, 0, stream>>>(w,    w_bf,    MROWS * DMODEL);
  k_cvt<<<4096, 256, 0, stream>>>(r,    r_bf,    RLEN * DMODEL);

  // 1) r projection (r_k + r_q[-1] bias)
  k_gemm_r<<<dim3((RLEN + 127) / 128, 2048 / 64), 128, 0, stream>>>(r_bf, wqkv_bf, rk, rq_last);

  // 2) QKV projection (Q gets +r_q[-1] bias; V stored transposed)
  k_gemm_qkv<<<dim3(MROWS / 128, NQKV / 64), 128, 0, stream>>>(w_bf, wqkv_bf, rq_last,
                                                               Qb, Kb, Vtb);

  // 3) relative attention (flash, rel-shift via LDS gather)
  size_t shmem = (size_t)2 * WLDSE * sizeof(__bf16);   // 2 waves per block
  k_attn<<<dim3(QLEN / 32, BSZ * NHEAD), 64, shmem, stream>>>(Qb, Kb, Vtb, rk, avb);

  // 4) output projection + residual
  k_gemm_out<<<dim3(MROWS / 128, DMODEL / 64), 128, 0, stream>>>(avb, wo_bf, w, xb);

  // 5) LayerNorm
  k_ln<<<MROWS, 256, 0, stream>>>(xb, gamma, beta, (float*)d_out);
}
